// MultiHeadAttention_71519795413768
// MI455X (gfx1250) — compile-verified, matrix operand order checked by reading, not run on hardware
//
#include <hip/hip_runtime.h>
#include <stdint.h>

typedef __bf16 bf16;
typedef __attribute__((ext_vector_type(16))) __bf16 v16bf;
typedef __attribute__((ext_vector_type(8)))  float  v8f;
typedef __attribute__((ext_vector_type(4)))  float  f4;

#define HEADS 16
#define DHEAD 64
#define TLEN  2048
#define BATCH 2
#define CDIM  1024

// 16-element bf16 fragment from an LDS row. Per CDNA5 16-bit A/B layout a lane
// holds K-chunks [k0..k0+7] and [k0+16..k0+23], k0 = (lane>>4)*8 (+32*c).
__device__ __forceinline__ v16bf load_frag(const bf16* row, int k0) {
  union { v16bf v; f4 f[2]; } u;
  u.f[0] = *(const f4*)(row + k0);
  u.f[1] = *(const f4*)(row + 16 + k0);
  return u.v;
}

__device__ __forceinline__ v8f wmma_bf16(v16bf a, v16bf b, v8f c) {
  return __builtin_amdgcn_wmma_f32_16x16x32_bf16(false, a, false, b, (short)0, c,
                                                 false, false);
}

// ---------------------------------------------------------------------------
// One-time fp32 -> bf16 conversion (8 elements per thread, vectorized).
// ---------------------------------------------------------------------------
__global__ void __launch_bounds__(256)
cvt_f32_bf16(const float* __restrict__ src, bf16* __restrict__ dst, int n8) {
  const int idx = blockIdx.x * 256 + threadIdx.x;
  if (idx >= n8) return;
  const f4 a = ((const f4*)src)[idx * 2 + 0];
  const f4 b = ((const f4*)src)[idx * 2 + 1];
  union { bf16 h[8]; f4 v; } u;
  u.h[0] = (bf16)a.x; u.h[1] = (bf16)a.y; u.h[2] = (bf16)a.z; u.h[3] = (bf16)a.w;
  u.h[4] = (bf16)b.x; u.h[5] = (bf16)b.y; u.h[6] = (bf16)b.z; u.h[7] = (bf16)b.w;
  ((f4*)dst)[idx] = u.v;
}

// ---------------------------------------------------------------------------
// GEMM: Y[M,N] (f32) = X[M,K] * W[N,K]^T, X/W bf16. Tile 128x64, BK=64.
// 8 waves, each a 32x32 patch (2x2 WMMA tiles); 8 WMMAs per barrier pair.
// ---------------------------------------------------------------------------
__global__ void __launch_bounds__(256)
gemm_xwt_wmma(const bf16* __restrict__ X, const bf16* __restrict__ W,
              float* __restrict__ Y, int M, int N, int K) {
  __shared__ __align__(16) bf16 As[128][64];
  __shared__ __align__(16) bf16 Bs[64][64];

  const int tid  = threadIdx.x;
  const int wave = tid >> 5;
  const int lane = tid & 31;
  const int half = lane >> 4;
  const int l16  = lane & 15;
  const int bm = blockIdx.y * 128;
  const int bn = blockIdx.x * 64;
  const int wm = (wave & 3) * 32;
  const int wn = (wave >> 2) * 32;

  v8f acc[2][2];
  acc[0][0] = {}; acc[0][1] = {}; acc[1][0] = {}; acc[1][1] = {};

  const int arow  = tid >> 1;          // 0..127
  const int acolb = (tid & 1) * 32;    // 32 elems / thread
  const int brow  = tid >> 2;          // 0..63
  const int bcolb = (tid & 3) * 16;    // 16 elems / thread
  const int k0 = half * 8;

  for (int kk = 0; kk < K; kk += 64) {
    {  // stage A: pure b128 copies, no conversion
      const f4* src = (const f4*)(X + (size_t)(bm + arow) * K + kk + acolb);
      f4* dst = (f4*)&As[arow][acolb];
#pragma unroll
      for (int c = 0; c < 4; c++) dst[c] = src[c];
    }
    {  // stage B
      const f4* src = (const f4*)(W + (size_t)(bn + brow) * K + kk + bcolb);
      f4* dst = (f4*)&Bs[brow][bcolb];
#pragma unroll
      for (int c = 0; c < 2; c++) dst[c] = src[c];
    }
    __syncthreads();

#pragma unroll
    for (int c = 0; c < 2; c++) {
      const int kc = c * 32 + k0;
      v16bf a0 = load_frag(&As[wm +      l16][0], kc);
      v16bf a1 = load_frag(&As[wm + 16 + l16][0], kc);
      v16bf b0 = load_frag(&Bs[wn +      l16][0], kc);
      v16bf b1 = load_frag(&Bs[wn + 16 + l16][0], kc);
      acc[0][0] = wmma_bf16(a0, b0, acc[0][0]);
      acc[0][1] = wmma_bf16(a0, b1, acc[0][1]);
      acc[1][0] = wmma_bf16(a1, b0, acc[1][0]);
      acc[1][1] = wmma_bf16(a1, b1, acc[1][1]);
    }
    __syncthreads();
  }

#pragma unroll
  for (int mt = 0; mt < 2; mt++)
#pragma unroll
    for (int nt = 0; nt < 2; nt++) {
      const int col = bn + wn + nt * 16 + l16;
#pragma unroll
      for (int r = 0; r < 8; r++) {
        const int row = bm + wm + mt * 16 + r + half * 8;
        Y[(size_t)row * N + col] = acc[mt][nt][r];
      }
    }
}

// ---------------------------------------------------------------------------
// RoPE + repack. Reads f32 qkv[B*T][3C]; writes head-major bf16 buffers
// Qb/Kb/Vb [(b*H+h)*T + t][64]. Softmax scale (1/8) folded into Qb.
// One thread per (row, head, pair i).
// ---------------------------------------------------------------------------
__global__ void __launch_bounds__(256)
rope_repack(const float* __restrict__ qkv, bf16* __restrict__ Qb,
            bf16* __restrict__ Kb, bf16* __restrict__ Vb) {
  const int idx = blockIdx.x * 256 + threadIdx.x;
  const int i   = idx & 31;
  const int h   = (idx >> 5) & (HEADS - 1);
  const int row = idx >> 9;               // b*T + t
  const int t   = row & (TLEN - 1);
  const int b   = row >> 11;              // T = 2048

  const float theta = __powf(10000.0f, -(float)i / 32.0f);
  float s, c;
  __sincosf((float)t * theta, &s, &c);

  const float* base = qkv + (size_t)row * (3 * CDIM) + h * (3 * DHEAD);
  const size_t dst = ((size_t)(b * HEADS + h) * TLEN + t) * DHEAD + 2 * i;

  union { bf16 h2[2]; uint32_t u; } p;
  const float x1 = base[2 * i], x2 = base[2 * i + 1];
  p.h2[0] = (bf16)((x1 * c - x2 * s) * 0.125f);
  p.h2[1] = (bf16)((x1 * s + x2 * c) * 0.125f);
  *(uint32_t*)(Qb + dst) = p.u;

  const float y1 = base[DHEAD + 2 * i], y2 = base[DHEAD + 2 * i + 1];
  p.h2[0] = (bf16)(y1 * c - y2 * s);
  p.h2[1] = (bf16)(y1 * s + y2 * c);
  *(uint32_t*)(Kb + dst) = p.u;

  p.h2[0] = (bf16)base[2 * DHEAD + 2 * i];
  p.h2[1] = (bf16)base[2 * DHEAD + 2 * i + 1];
  *(uint32_t*)(Vb + dst) = p.u;
}

// ---------------------------------------------------------------------------
// Sliding-window flash attention on bf16 head-major Q/K/V. One workgroup =
// (b,h,128 queries); 8 waves x 16 queries. K tiles staged via CDNA5 async
// global->LDS (ASYNCcnt), V transposed in LDS for contiguous PV B-fragments.
// Output bf16 att[B*T][C] (transpose(0,2,1,3) layout built into addressing).
// ---------------------------------------------------------------------------
__global__ void __launch_bounds__(256)
attn_kernel(const bf16* __restrict__ Qb, const bf16* __restrict__ Kb,
            const bf16* __restrict__ Vb, bf16* __restrict__ att,
            const int* __restrict__ window_p) {
  __shared__ __align__(16) bf16 Klds[32][64];
  __shared__ __align__(16) bf16 Vlds[64][32];
  __shared__ __align__(16) bf16 Plds[8][16][32];

  const int window = *window_p;
  const int tid  = threadIdx.x;
  const int wave = tid >> 5;
  const int lane = tid & 31;
  const int half = lane >> 4;
  const int l16  = lane & 15;

  const int nqb   = TLEN / 128;
  const int bh    = blockIdx.x / nqb;      // b*HEADS + h
  const int b     = bh / HEADS;
  const int h     = bh % HEADS;
  const int qbase = (blockIdx.x % nqb) * 128;
  const int qw    = qbase + wave * 16;
  const size_t headbase = (size_t)bh * TLEN * DHEAD;
  const int k0 = half * 8;

  // --- Q fragments (already scaled), contiguous vector loads ---
  v16bf qf[2];
  {
    const bf16* qsrc = Qb + headbase + (size_t)(qw + l16) * DHEAD;
#pragma unroll
    for (int kc = 0; kc < 2; kc++) qf[kc] = load_frag(qsrc, kc * 32 + k0);
  }

  v8f o[4];
  o[0] = {}; o[1] = {}; o[2] = {}; o[3] = {};
  float mmax[8], lsum[8];
#pragma unroll
  for (int r = 0; r < 8; r++) { mmax[r] = -1e30f; lsum[r] = 0.0f; }

  const int kfirst = (qbase - window) & ~31;
  const int kend   = qbase + 128;

  for (int kb = kfirst; kb < kend; kb += 32) {
    // ---- stage K (async global->LDS) and V (transposed) ----
    {
      const int kk   = tid >> 3;        // key in tile
      const int dseg = (tid & 7) * 8;   // 8 d-elems per thread
      const int j = kb + kk;
      if (j >= 0 && j < TLEN) {
        const bf16* ksrc = Kb + headbase + (size_t)j * DHEAD + dseg;
        const uint32_t ldsoff = (uint32_t)(uintptr_t)&Klds[kk][dseg];
        asm volatile("global_load_async_to_lds_b128 %0, %1, off"
                     :: "v"(ldsoff), "v"(ksrc) : "memory");
        union { f4 v; bf16 e[8]; } uv;
        uv.v = *(const f4*)(Vb + headbase + (size_t)j * DHEAD + dseg);
#pragma unroll
        for (int e = 0; e < 8; e++) Vlds[dseg + e][kk] = uv.e[e];
      } else {
        union { bf16 z[8]; f4 v; } uz;
#pragma unroll
        for (int e = 0; e < 8; e++) uz.z[e] = (bf16)0.0f;
        *(f4*)&Klds[kk][dseg] = uz.v;
#pragma unroll
        for (int e = 0; e < 8; e++) Vlds[dseg + e][kk] = (bf16)0.0f;
      }
    }
    asm volatile("s_wait_asynccnt 0" ::: "memory");
    __syncthreads();

    if (kb + 31 >= qw - window && kb <= qw + 15) {
      // ---- S = Q K^T : two 16x16 tiles over 32 keys ----
      v8f s[2];
#pragma unroll
      for (int nt = 0; nt < 2; nt++) {
        const bf16* krow = &Klds[nt * 16 + l16][0];
        v8f a = {};
        a = wmma_bf16(qf[0], load_frag(krow, k0), a);
        a = wmma_bf16(qf[1], load_frag(krow + 32, k0), a);
        s[nt] = a;
      }

      // ---- mask + online softmax (rows live in 16-lane halves) ----
      float tmax[8];
#pragma unroll
      for (int r = 0; r < 8; r++) {
        const int i_q = qw + r + half * 8;
#pragma unroll
        for (int nt = 0; nt < 2; nt++) {
          const int j = kb + nt * 16 + l16;
          const bool valid = (j >= 0) && (j <= i_q) && (j >= i_q - window);
          if (!valid) s[nt][r] = -1e30f;
        }
        float v = fmaxf(s[0][r], s[1][r]);
        v = fmaxf(v, __shfl_xor(v, 1, 32));
        v = fmaxf(v, __shfl_xor(v, 2, 32));
        v = fmaxf(v, __shfl_xor(v, 4, 32));
        v = fmaxf(v, __shfl_xor(v, 8, 32));
        tmax[r] = v;
      }
#pragma unroll
      for (int r = 0; r < 8; r++) {
        const float mnew = fmaxf(mmax[r], tmax[r]);
        const float corr = __expf(mmax[r] - mnew);
        mmax[r] = mnew;
        lsum[r] *= corr;
#pragma unroll
        for (int dt = 0; dt < 4; dt++) o[dt][r] *= corr;
        const float p0 = __expf(s[0][r] - mnew);
        const float p1 = __expf(s[1][r] - mnew);
        s[0][r] = p0;
        s[1][r] = p1;
        float ps = p0 + p1;
        ps += __shfl_xor(ps, 1, 32);
        ps += __shfl_xor(ps, 2, 32);
        ps += __shfl_xor(ps, 4, 32);
        ps += __shfl_xor(ps, 8, 32);
        lsum[r] += ps;
      }

      // ---- C-layout P -> A-layout bf16 via per-wave LDS scratch ----
#pragma unroll
      for (int r = 0; r < 8; r++) {
        const int m = r + half * 8;
        Plds[wave][m][l16]      = (bf16)s[0][r];
        Plds[wave][m][16 + l16] = (bf16)s[1][r];
      }
      asm volatile("s_wait_dscnt 0" ::: "memory");
      v16bf pf = load_frag(&Plds[wave][l16][0], k0);

      // ---- O += P V ----
#pragma unroll
      for (int dt = 0; dt < 4; dt++) {
        v16bf vf = load_frag(&Vlds[dt * 16 + l16][0], k0);
        o[dt] = wmma_bf16(pf, vf, o[dt]);
      }
    }
    __syncthreads();
  }

  // ---- normalize, store bf16 att[b*T+q][h*64+d] ----
#pragma unroll
  for (int r = 0; r < 8; r++) {
    const float inv = 1.0f / lsum[r];
    const size_t row = (size_t)(b * TLEN + qw + r + half * 8);
#pragma unroll
    for (int dt = 0; dt < 4; dt++) {
      att[row * CDIM + h * DHEAD + dt * 16 + l16] = (bf16)(o[dt][r] * inv);
    }
  }
}

// ---------------------------------------------------------------------------
extern "C" void kernel_launch(void* const* d_in, const int* in_sizes, int n_in,
                              void* d_out, int out_size, void* d_ws,
                              size_t ws_size, hipStream_t stream) {
  (void)in_sizes; (void)n_in; (void)out_size; (void)ws_size;
  const float* x     = (const float*)d_in[0];
  const float* w_qkv = (const float*)d_in[1];
  const float* w_out = (const float*)d_in[2];
  const int*   win   = (const int*)d_in[3];
  float* out = (float*)d_out;

  const size_t M    = (size_t)BATCH * TLEN;       // 4096
  const size_t NX   = M * CDIM;                   // 4,194,304
  const size_t NWQ  = (size_t)3 * CDIM * CDIM;    // 3,145,728
  const size_t NWO  = (size_t)CDIM * CDIM;        // 1,048,576
  const size_t NHTD = (size_t)BATCH * HEADS * TLEN * DHEAD;  // 4,194,304

  char* w = (char*)d_ws;
  bf16* xb    = (bf16*)w; w += NX * sizeof(bf16);
  bf16* wqkvb = (bf16*)w; w += NWQ * sizeof(bf16);
  bf16* woutb = (bf16*)w; w += NWO * sizeof(bf16);
  bf16* qb    = (bf16*)w; w += NHTD * sizeof(bf16);
  bf16* kb    = (bf16*)w; w += NHTD * sizeof(bf16);
  bf16* vb    = (bf16*)w; w += NHTD * sizeof(bf16);
  bf16* attb  = (bf16*)w; w += NX * sizeof(bf16);
  float* qkv  = (float*)w;                        // 4096 x 3072 f32

  dim3 blk(256);

  // 0) one-time fp32 -> bf16 packs
  cvt_f32_bf16<<<dim3((int)(NX / 8 / 256)),  blk, 0, stream>>>(x, xb, (int)(NX / 8));
  cvt_f32_bf16<<<dim3((int)(NWQ / 8 / 256)), blk, 0, stream>>>(w_qkv, wqkvb, (int)(NWQ / 8));
  cvt_f32_bf16<<<dim3((int)(NWO / 8 / 256)), blk, 0, stream>>>(w_out, woutb, (int)(NWO / 8));

  // 1) QKV projection: qkv[4096,3072] = xb * wqkvb^T
  gemm_xwt_wmma<<<dim3(3 * CDIM / 64, (int)M / 128), blk, 0, stream>>>(
      xb, wqkvb, qkv, (int)M, 3 * CDIM, CDIM);

  // 2) RoPE + head-major bf16 repack (scale folded into Q)
  rope_repack<<<dim3((int)(M * HEADS * 32 / 256)), blk, 0, stream>>>(qkv, qb, kb, vb);

  // 3) sliding-window attention -> attb (bf16, [b,t][h*64+d])
  attn_kernel<<<dim3(BATCH * HEADS * (TLEN / 128)), blk, 0, stream>>>(
      qb, kb, vb, attb, win);

  // 4) output projection: out[4096,1024] = attb * woutb^T
  gemm_xwt_wmma<<<dim3(CDIM / 64, (int)M / 128), blk, 0, stream>>>(
      attb, woutb, out, (int)M, CDIM, CDIM);
}
